// PointNet2Encoder_44770739093792
// MI455X (gfx1250) — matrix operand order repaired and checked by reading, not compiled, not run on hardware
//
#include <hip/hip_runtime.h>

typedef _Float16 half_t;
typedef __attribute__((ext_vector_type(16))) half_t v16h;
typedef __attribute__((ext_vector_type(8)))  float  v8f;

__device__ __forceinline__ v8f v8f_zero() {
  v8f z = {0.f, 0.f, 0.f, 0.f, 0.f, 0.f, 0.f, 0.f};
  return z;
}

// ---------------------------------------------------------------------------
// Furthest point sampling. One block (256 threads) per batch. Coords + running
// min-dist live in registers (NPT points per thread); argmax via LDS tree.
// Seeded at index 0, matching the reference.
// ---------------------------------------------------------------------------
template<int N, int M, int NPT>
__global__ __launch_bounds__(256) void fps_kernel(const float* __restrict__ pts,
                                                  float* __restrict__ prop) {
  const int b = blockIdx.x;
  const int tid = threadIdx.x;
  __shared__ float bpt[3];
  __shared__ float rv[256];
  __shared__ int   ri[256];

  const float* px = pts + (size_t)b * 3 * N;
  const float* py = px + N;
  const float* pz = py + N;

  float lx[NPT], ly[NPT], lz[NPT], ld[NPT];
#pragma unroll
  for (int j = 0; j < NPT; ++j) {
    const int i = tid * NPT + j;
    lx[j] = px[i]; ly[j] = py[i]; lz[j] = pz[i];
    ld[j] = 1e30f;
  }

  float* ox = prop + (size_t)b * 3 * M;
  float* oy = ox + M;
  float* oz = oy + M;

  if (tid == 0) {
    bpt[0] = lx[0]; bpt[1] = ly[0]; bpt[2] = lz[0];
    ox[0] = lx[0]; oy[0] = ly[0]; oz[0] = lz[0];
  }
  __syncthreads();

  for (int it = 1; it < M; ++it) {
    const float bx = bpt[0], by = bpt[1], bz = bpt[2];
    float bestv = -1.f;
    int   besti = tid * NPT;
#pragma unroll
    for (int j = 0; j < NPT; ++j) {
      const float dx = lx[j] - bx, dy = ly[j] - by, dz = lz[j] - bz;
      float d = dx * dx + dy * dy + dz * dz;
      d = fminf(ld[j], d);
      ld[j] = d;
      if (d > bestv) { bestv = d; besti = tid * NPT + j; }
    }
    rv[tid] = bestv; ri[tid] = besti;
    __syncthreads();
    for (int s = 128; s > 0; s >>= 1) {
      if (tid < s) {
        const float v2 = rv[tid + s];
        const int   i2 = ri[tid + s];
        if (v2 > rv[tid] || (v2 == rv[tid] && i2 < ri[tid])) {
          rv[tid] = v2; ri[tid] = i2;
        }
      }
      __syncthreads();
    }
    const int widx = ri[0];
    if (tid == widx / NPT) {
      const int sl = widx % NPT;
      bpt[0] = lx[sl]; bpt[1] = ly[sl]; bpt[2] = lz[sl];
      ox[it] = lx[sl]; oy[it] = ly[sl]; oz[it] = lz[sl];
    }
    __syncthreads();
  }
}

// ---------------------------------------------------------------------------
// KNN: one block per (query, batch). Full distance row in LDS, then K rounds
// of argmin-extract (ascending distance, tie -> lower index == top_k(-d)).
// ---------------------------------------------------------------------------
template<int N, int M, int K>
__global__ __launch_bounds__(256) void knn_kernel(const float* __restrict__ pts,
                                                  const float* __restrict__ qpts,
                                                  int* __restrict__ gidx) {
  const int b = blockIdx.y, m = blockIdx.x, tid = threadIdx.x;
  __shared__ float dist[N];
  __shared__ float rv[256];
  __shared__ int   ri[256];

  const float* px = pts + (size_t)b * 3 * N;
  const float* py = px + N;
  const float* pz = py + N;
  const float qx = qpts[(size_t)b * 3 * M + m];
  const float qy = qpts[(size_t)b * 3 * M + M + m];
  const float qz = qpts[(size_t)b * 3 * M + 2 * M + m];

  for (int i = tid; i < N; i += 256) {
    const float dx = px[i] - qx, dy = py[i] - qy, dz = pz[i] - qz;
    dist[i] = dx * dx + dy * dy + dz * dz;
  }
  __syncthreads();

  int* outp = gidx + ((size_t)b * M + m) * K;
  for (int k = 0; k < K; ++k) {
    float bv = 3e38f;
    int   bi = 0;
    for (int i = tid; i < N; i += 256) {
      const float d = dist[i];
      if (d < bv || (d == bv && i < bi)) { bv = d; bi = i; }
    }
    rv[tid] = bv; ri[tid] = bi;
    __syncthreads();
    for (int s = 128; s > 0; s >>= 1) {
      if (tid < s) {
        const float v2 = rv[tid + s];
        const int   i2 = ri[tid + s];
        if (v2 < rv[tid] || (v2 == rv[tid] && i2 < ri[tid])) {
          rv[tid] = v2; ri[tid] = i2;
        }
      }
      __syncthreads();
    }
    if (tid == 0) {
      outp[k] = ri[0];
      dist[ri[0]] = 3e38f;
    }
    __syncthreads();
  }
}

// ---------------------------------------------------------------------------
// Weight pre-conversion: f32 [COUT, CACT] -> padded f16 [COUT, CINP] (zeros in
// the padded channel tail). Done once per call; weights then stay L2-resident
// and A-fragments are loaded straight from global by every MLP block.
// ---------------------------------------------------------------------------
__global__ __launch_bounds__(256) void cvt_w_kernel(const float* __restrict__ W,
                                                    half_t* __restrict__ outp,
                                                    int cout, int cact, int cinp) {
  const int idx = blockIdx.x * 256 + threadIdx.x;
  if (idx >= cout * cinp) return;
  const int o = idx / cinp;
  const int k = idx - o * cinp;
  outp[idx] = (k < cact) ? (half_t)W[(size_t)o * cact + k] : (half_t)0.f;
}

// ---------------------------------------------------------------------------
// WMMA fragment helpers (wave32, V_WMMA_F32_16X16X32_F16; A/B use this code's contiguous k order)
//   A 16x32 f16 : lane L (row = L&15): elem j -> K = 16*(L>=16) + j
//
//   B 32x16 f16 : lane L (col = L&15): elem j -> K = 16*(L>=16) + j
//   C/D 16x16 f32: lane L (col = L&15): elem r -> row = r + 8*(L>=16)
//
// Activations live in LDS column-major [64][S] with S = CIN + 16 halfs of skew:
//  - per-lane B fragment = 16 contiguous halves at a 32B-aligned offset
//    -> merges into 2x ds_load_b128
//  - per-lane conv output = 8 contiguous halves at a 16B-aligned offset
//    -> merges into 1x ds_store_b128
//  - column byte stride (2*S) is not a multiple of the 256B bank rotation
// ---------------------------------------------------------------------------
// A fragment loaded directly from global f16 weights (two contiguous 16-byte
// runs per lane -> two global_load_b128, L2/WGP$-resident after warmup).
__device__ __forceinline__ v16h load_a_gmem(const half_t* __restrict__ wr, int hi) {
  v16h a;
#pragma unroll
  for (int j = 0; j < 8; ++j) a[j] = wr[hi * 16 + j];
#pragma unroll
  for (int j = 0; j < 8; ++j) a[8 + j] = wr[hi * 16 + 8 + j];
  return a;
}

// B fragment from LDS activations, column-major with stride S halfs.
template<int S>
__device__ __forceinline__ v16h load_b_cm(const half_t* __restrict__ xin,
                                          int kc, int nt, int hi, int cl) {
  const half_t* xc = xin + (size_t)(nt * 16 + cl) * S + kc * 32 + hi * 16;
  v16h bf;
#pragma unroll
  for (int j = 0; j < 16; ++j) bf[j] = xc[j];
  return bf;
}

// One 1x1 conv layer (COUT x CINC GEMM over 64 columns), scale+bias+ReLU,
// f16 result written column-major to LDS xout (stride COUT+16).
// Weights wp: padded f16 [COUT][CINC]. No barriers inside the K-loop.
template<int COUT, int CINC>
__device__ __forceinline__ void conv_lds(const half_t* __restrict__ xin,
                                         half_t* __restrict__ xout,
                                         const half_t* __restrict__ wp,
                                         const float* __restrict__ G,
                                         const float* __restrict__ Bv) {
  constexpr int SIN  = CINC + 16;
  constexpr int SOUT = COUT + 16;
  constexpr int ROWT = COUT / 16;
  constexpr int PPW  = (ROWT * 4) / 8;  // output tiles per wave (8 waves)
  const int tid = threadIdx.x;
  const int w = tid >> 5, lane = tid & 31, hi = lane >> 4, cl = lane & 15;
  const int nt = w & 3;        // constant n-tile per wave
  const int mt0 = w >> 2;      // row-tiles: mt0, mt0+2, mt0+4, ...

  v8f acc[PPW];
#pragma unroll
  for (int i = 0; i < PPW; ++i) acc[i] = v8f_zero();

#pragma unroll
  for (int kc = 0; kc < CINC / 32; ++kc) {
    const v16h bf = load_b_cm<SIN>(xin, kc, nt, hi, cl);  // shared by all tiles
#pragma unroll
    for (int i = 0; i < PPW; ++i) {
      const int mt = mt0 + i * 2;
      const v16h a = load_a_gmem(wp + (size_t)(mt * 16 + cl) * CINC + kc * 32, hi);
      acc[i] = __builtin_amdgcn_wmma_f32_16x16x32_f16(
          false, a, false, bf, (short)0, acc[i], false, false);
    }
  }

#pragma unroll
  for (int i = 0; i < PPW; ++i) {
    const int mt = mt0 + i * 2;
    const int col = nt * 16 + cl;
    half_t* xo = xout + (size_t)col * SOUT + mt * 16 + hi * 8;  // 16B aligned
#pragma unroll
    for (int r = 0; r < 8; ++r) {
      const int ch = mt * 16 + hi * 8 + r;
      const float v = fmaxf(0.f, acc[i][r] * G[ch] + Bv[ch]);
      xo[r] = (half_t)v;  // 8 contiguous halves -> ds_store_b128
    }
  }
  // Single barrier per conv boundary: makes xout visible to the next conv and
  // guarantees this conv's xin reads are done before the next conv overwrites.
  __syncthreads();
}

// Final conv layer + max-pool over KS samples -> global feats [B, COUT, MQ].
template<int COUT, int CINC, int KS>
__device__ __forceinline__ void conv_pool(const half_t* __restrict__ xin,
                                          const half_t* __restrict__ wp,
                                          const float* __restrict__ G,
                                          const float* __restrict__ Bv,
                                          float* __restrict__ outp,
                                          int b, int q0, int MQ) {
  constexpr int SIN  = CINC + 16;
  constexpr int ROWT = COUT / 16;
  constexpr int MTW  = ROWT / 8;   // row-tiles owned per wave
  constexpr int GN   = 64 / KS;    // pooling groups (queries) per tile
  constexpr int NTG  = 4 / GN;     // n-tiles per group
  const int tid = threadIdx.x;
  const int w = tid >> 5, lane = tid & 31, hi = lane >> 4, cl = lane & 15;

  v8f acc[MTW][4];
#pragma unroll
  for (int i = 0; i < MTW; ++i)
#pragma unroll
    for (int nt = 0; nt < 4; ++nt) acc[i][nt] = v8f_zero();

#pragma unroll
  for (int kc = 0; kc < CINC / 32; ++kc) {
    v16h bf[4];
#pragma unroll
    for (int nt = 0; nt < 4; ++nt) bf[nt] = load_b_cm<SIN>(xin, kc, nt, hi, cl);
#pragma unroll
    for (int i = 0; i < MTW; ++i) {
      const int mt = w + i * 8;
      const v16h a = load_a_gmem(wp + (size_t)(mt * 16 + cl) * CINC + kc * 32, hi);
#pragma unroll
      for (int nt = 0; nt < 4; ++nt) {
        acc[i][nt] = __builtin_amdgcn_wmma_f32_16x16x32_f16(
            false, a, false, bf[nt], (short)0, acc[i][nt], false, false);
      }
    }
  }

#pragma unroll
  for (int i = 0; i < MTW; ++i) {
    const int mt = w + i * 8;
#pragma unroll
    for (int r = 0; r < 8; ++r) {
      const int ch = mt * 16 + hi * 8 + r;
      const float gg = G[ch], bb = Bv[ch];
#pragma unroll
      for (int g = 0; g < GN; ++g) {
        float mv = 0.f;  // all candidates post-ReLU are >= 0
#pragma unroll
        for (int t = 0; t < NTG; ++t) {
          const int nt = g * NTG + t;
          mv = fmaxf(mv, fmaxf(0.f, acc[i][nt][r] * gg + bb));
        }
        // max over the 16 columns held by this half-wave
        mv = fmaxf(mv, __shfl_xor(mv, 1, 32));
        mv = fmaxf(mv, __shfl_xor(mv, 2, 32));
        mv = fmaxf(mv, __shfl_xor(mv, 4, 32));
        mv = fmaxf(mv, __shfl_xor(mv, 8, 32));
        if (cl == 0)
          outp[((size_t)b * COUT + ch) * MQ + (q0 + g)] = mv;
      }
    }
  }
}

// ---------------------------------------------------------------------------
// Fused gather + 3x(1x1 conv,BN,ReLU) + max-pool. One block = 64 columns
// (= 64/KS queries x KS samples). Activations ping-pong in LDS (column-major,
// skewed); weights come from pre-converted f16 global arrays (L2-resident).
// ---------------------------------------------------------------------------
template<int CINP, int C1, int C2, int C3, int KS, int CF, int NPTS, int MQ>
__global__ __launch_bounds__(256) void mlp_kernel(
    const float* __restrict__ points, const float* __restrict__ prop,
    const float* __restrict__ feats, const int* __restrict__ gidx,
    const half_t* __restrict__ wp0, const float* __restrict__ G0, const float* __restrict__ B0,
    const half_t* __restrict__ wp1, const float* __restrict__ G1, const float* __restrict__ B1,
    const half_t* __restrict__ wp2, const float* __restrict__ G2, const float* __restrict__ B2,
    float* __restrict__ fout) {
  constexpr int SA = ((CINP > C2) ? CINP : C2) + 16;  // bufA stride capacity
  constexpr int SB = C1 + 16;
  __shared__ half_t bufA[64 * SA];
  __shared__ half_t bufB[64 * SB];
  __shared__ int    sidx[64];
  __shared__ float  scent[3 * 64];

  const int b = blockIdx.y;
  constexpr int QPB = 64 / KS;
  const int q0 = blockIdx.x * QPB;
  const int tid = threadIdx.x;

  if (tid == 0) {
    __builtin_prefetch(wp0, 0, 1);
    __builtin_prefetch(wp1, 0, 1);
    __builtin_prefetch(wp2, 0, 1);
  }

  if (tid < 64) {
    const int q = q0 + tid / KS;
    const int s = tid % KS;
    sidx[tid] = gidx[((size_t)b * MQ + q) * KS + s];
#pragma unroll
    for (int c = 0; c < 3; ++c)
      scent[c * 64 + tid] = prop[((size_t)b * 3 + c) * MQ + q];
  }
  __syncthreads();

  // Gather + concat + pad into bufA, column-major [64][CINP+16]
  constexpr int SIN0 = CINP + 16;
  for (int t = tid; t < CINP * 64; t += 256) {
    const int ch = t >> 6, col = t & 63;
    const int id = sidx[col];
    float v = 0.f;
    if (ch < 3)
      v = points[((size_t)b * 3 + ch) * NPTS + id] - scent[ch * 64 + col];
    else if (ch < 3 + CF)
      v = feats[((size_t)b * CF + (ch - 3)) * NPTS + id];
    bufA[(size_t)col * SIN0 + ch] = (half_t)v;
  }
  __syncthreads();

  conv_lds<C1, CINP>(bufA, bufB, wp0, G0, B0);
  conv_lds<C2, C1>(bufB, bufA, wp1, G1, B1);
  conv_pool<C3, C2, KS>(bufA, wp2, G2, B2, fout, b, q0, MQ);
}

// ---------------------------------------------------------------------------
extern "C" void kernel_launch(void* const* d_in, const int* in_sizes, int n_in,
                              void* d_out, int out_size, void* d_ws, size_t ws_size,
                              hipStream_t stream) {
  (void)in_sizes; (void)n_in; (void)out_size; (void)ws_size;
  const float* points = (const float*)d_in[0];
  const float* w10 = (const float*)d_in[1];
  const float* g10 = (const float*)d_in[2];
  const float* b10 = (const float*)d_in[3];
  const float* w11 = (const float*)d_in[4];
  const float* g11 = (const float*)d_in[5];
  const float* b11 = (const float*)d_in[6];
  const float* w12 = (const float*)d_in[7];
  const float* g12 = (const float*)d_in[8];
  const float* b12 = (const float*)d_in[9];
  const float* w20 = (const float*)d_in[10];
  const float* g20 = (const float*)d_in[11];
  const float* b20 = (const float*)d_in[12];
  const float* w21 = (const float*)d_in[13];
  const float* g21 = (const float*)d_in[14];
  const float* b21 = (const float*)d_in[15];
  const float* w22 = (const float*)d_in[16];
  const float* g22 = (const float*)d_in[17];
  const float* b22 = (const float*)d_in[18];

  float* out    = (float*)d_out;
  float* prop1  = out;                                // [16,3,1024]
  float* feats1 = prop1 + (size_t)16 * 3 * 1024;      // [16,128,1024]
  float* prop2  = feats1 + (size_t)16 * 128 * 1024;   // [16,3,256]
  float* feats2 = prop2 + (size_t)16 * 3 * 256;       // [16,256,256]

  int* gidx1 = (int*)d_ws;                            // [16,1024,32]
  int* gidx2 = gidx1 + (size_t)16 * 1024 * 32;        // [16,256,64]

  // Pre-converted, channel-padded f16 weights in workspace.
  half_t* wp10 = (half_t*)(gidx2 + (size_t)16 * 256 * 64);
  half_t* wp11 = wp10 + (size_t)64 * 32;    // [64][32]   (6 -> 32)
  half_t* wp12 = wp11 + (size_t)64 * 64;    // [64][64]
  half_t* wp20 = wp12 + (size_t)128 * 64;   // [128][64]
  half_t* wp21 = wp20 + (size_t)128 * 160;  // [128][160] (131 -> 160)
  half_t* wp22 = wp21 + (size_t)128 * 128;  // [128][128]
                                            // [256][128]

  cvt_w_kernel<<<(64 * 32 + 255) / 256, 256, 0, stream>>>(w10, wp10, 64, 6, 32);
  cvt_w_kernel<<<(64 * 64 + 255) / 256, 256, 0, stream>>>(w11, wp11, 64, 64, 64);
  cvt_w_kernel<<<(128 * 64 + 255) / 256, 256, 0, stream>>>(w12, wp12, 128, 64, 64);
  cvt_w_kernel<<<(128 * 160 + 255) / 256, 256, 0, stream>>>(w20, wp20, 128, 131, 160);
  cvt_w_kernel<<<(128 * 128 + 255) / 256, 256, 0, stream>>>(w21, wp21, 128, 128, 128);
  cvt_w_kernel<<<(256 * 128 + 255) / 256, 256, 0, stream>>>(w22, wp22, 256, 128, 128);

  // ----- Layer 1 -----
  fps_kernel<4096, 1024, 16><<<16, 256, 0, stream>>>(points, prop1);
  knn_kernel<4096, 1024, 32><<<dim3(1024, 16), 256, 0, stream>>>(points, prop1, gidx1);
  mlp_kernel<32, 64, 64, 128, 32, 3, 4096, 1024>
      <<<dim3(512, 16), 256, 0, stream>>>(points, prop1, points, gidx1,
                                          wp10, g10, b10, wp11, g11, b11,
                                          wp12, g12, b12, feats1);

  // ----- Layer 2 -----
  fps_kernel<1024, 256, 4><<<16, 256, 0, stream>>>(prop1, prop2);
  knn_kernel<1024, 256, 64><<<dim3(256, 16), 256, 0, stream>>>(prop1, prop2, gidx2);
  mlp_kernel<160, 128, 128, 256, 64, 128, 1024, 256>
      <<<dim3(256, 16), 256, 0, stream>>>(prop1, prop2, feats1, gidx2,
                                          wp20, g20, b20, wp21, g21, b21,
                                          wp22, g22, b22, feats2);
}
